// PGCN_89859305767455
// MI455X (gfx1250) — matrix operand order repaired and checked by reading, not compiled, not run on hardware
//
#include <hip/hip_runtime.h>

// ---------------------------------------------------------------------------
// PGCN pipeline for MI455X (gfx1250, wave32, WMMA).
// Heavy matmuls -> v_wmma_f32_16x16x32_bf16, operands DMA'd into LDS with
// GLOBAL_LOAD_ASYNC_TO_LDS_B128 (ASYNCcnt), double-buffered, last-iteration
// peeled so the steady-state K loop is branch-free.
// ---------------------------------------------------------------------------

typedef __bf16 bf16;
typedef __attribute__((ext_vector_type(16))) __bf16        v16bf;
typedef __attribute__((ext_vector_type(8)))  float         v8f;
typedef __attribute__((ext_vector_type(8)))  unsigned short v8us;

#define TM 128
#define TN 128
#define TK 32
#define LDK 40   // padded halves per LDS row (32 data + 8 pad) -> fewer bank conflicts

union Frag { v8us u[2]; v16bf v; };

static __device__ __forceinline__ unsigned short f2bf_bits(float f) {
  bf16 h = (bf16)f;
  return __builtin_bit_cast(unsigned short, h);
}

// Async global->LDS 16B copy (per lane), tracked by ASYNCcnt.
static __device__ __forceinline__ void async_copy_b128(unsigned lds_off, const void* gptr) {
  asm volatile("global_load_async_to_lds_b128 %0, %1, off"
               :: "v"(lds_off), "v"(gptr)
               : "memory");
}
template <int N>
static __device__ __forceinline__ void wait_asynccnt() {
  asm volatile("s_wait_asynccnt %0" :: "n"(N) : "memory");
}

// Epilogue modes
#define EPI_COS   0  // c /= (nrm[m]*nrm[n]+1e-6); store f32 row-major
#define EPI_TBF   1  // store bf16 TRANSPOSED  Ct[n][m]  (contiguous 16B lane stores)
#define EPI_BRELU 2  // c = relu(c + bias[n]); store bf16 row-major
#define EPI_BRES  3  // c = c + bias[n] + res[m][n]; store bf16 row-major
#define EPI_HEAD  4  // if n < nlim: store f32 (c + bias[n]) row-major (ldc = 81)

// Generic GEMM: C[m,n] = sum_k A[m,k] * Bn[n,k]   (Bn is B transposed, K-contiguous)
// Requirements: M % 128 == 0, N % 128 == 0, K % 32 == 0 (callers guarantee).
template <int EPI>
__global__ __launch_bounds__(256) void gemm_bf16_wmma(
    const bf16* __restrict__ A, long long strideA, int lda,
    const bf16* __restrict__ Bn, long long strideB, int ldb,
    int M, int N, int K,
    float* __restrict__ Cf, long long strideCf, int ldc,
    bf16* __restrict__ Cb, long long strideCb, int ldcb,
    const float* __restrict__ bias,
    const float* __restrict__ res, long long strideRes, int ldres,
    const float* __restrict__ nrm, long long strideNrm,
    int nlim) {
  __shared__ __align__(16) unsigned short shA[2][TM * LDK];
  __shared__ __align__(16) unsigned short shB[2][TN * LDK];

  const int b = blockIdx.z;
  A += (long long)b * strideA;
  Bn += (long long)b * strideB;
  if (Cf)  Cf += (long long)b * strideCf;
  if (Cb)  Cb += (long long)b * strideCb;
  if (res) res += (long long)b * strideRes;
  if (nrm) nrm += (long long)b * strideNrm;

  const int tid = threadIdx.x;
  const int tileM = blockIdx.y * TM;
  const int tileN = blockIdx.x * TN;

  // Each thread owns 2 x 16B chunks of the A tile and 2 of the B tile.
  int arow[2], acol[2];
#pragma unroll
  for (int i = 0; i < 2; ++i) {
    int c = tid + 256 * i;          // 512 chunks cover 128 rows x 32 halves
    arow[i] = c >> 2;               // 4 chunks (of 8 halves) per row
    acol[i] = (c & 3) * 8;          // half offset within row
  }

  // Per-thread global pointers, bumped by TK after each issued tile.
  const bf16* gA[2];
  const bf16* gB[2];
#pragma unroll
  for (int i = 0; i < 2; ++i) {
    gA[i] = A + (size_t)(tileM + arow[i]) * lda + acol[i];
    gB[i] = Bn + (size_t)(tileN + arow[i]) * ldb + acol[i];
  }
  // LDS byte offsets (low 32 bits of generic pointer == LDS offset).
  unsigned ldsA[2], ldsB[2];
#pragma unroll
  for (int i = 0; i < 2; ++i) {
    ldsA[i] = (unsigned)(uintptr_t)&shA[0][arow[i] * LDK + acol[i]];
    ldsB[i] = (unsigned)(uintptr_t)&shB[0][arow[i] * LDK + acol[i]];
  }
  const unsigned bufStrideA = (unsigned)(TM * LDK * 2);
  const unsigned bufStrideB = (unsigned)(TN * LDK * 2);

  // Issue one K-tile of async copies into LDS buffer `buf`, advance pointers.
  auto issue_tile = [&](int buf) {
    const unsigned oa = (unsigned)buf * bufStrideA;
    const unsigned ob = (unsigned)buf * bufStrideB;
#pragma unroll
    for (int i = 0; i < 2; ++i) {
      async_copy_b128(ldsA[i] + oa, gA[i]);
      gA[i] += TK;
    }
#pragma unroll
    for (int i = 0; i < 2; ++i) {
      async_copy_b128(ldsB[i] + ob, gB[i]);
      gB[i] += TK;
    }
  };

  const int lane = tid & 31;
  const int wid  = tid >> 5;
  const int wm   = (wid & 3) * 32;  // wave M offset (4 waves along M)
  const int wn   = (wid >> 2) * 64; // wave N offset (2 waves along N)
  const int lrow = lane & 15;
  const int lh   = lane >> 4;

  v8f acc[2][4];
#pragma unroll
  for (int i = 0; i < 2; ++i)
#pragma unroll
    for (int j = 0; j < 4; ++j) {
      v8f z = {0.f, 0.f, 0.f, 0.f, 0.f, 0.f, 0.f, 0.f};
      acc[i][j] = z;
    }

  // 8 WMMAs on LDS buffer `buf`.
  auto compute_tile = [&](int buf) {
    // A fragment (16x32 bf16): lanes 0-15 rows m, halves {k 0-7, k 16-23};
    //                          lanes 16-31 same rows, halves {k 8-15, k 24-31}.
    Frag fa[2];
#pragma unroll
    for (int i = 0; i < 2; ++i) {
      int r = wm + i * 16 + lrow;
      fa[i].u[0] = *reinterpret_cast<const v8us*>(&shA[buf][r * LDK + lh * 8]);
      fa[i].u[1] = *reinterpret_cast<const v8us*>(&shA[buf][r * LDK + 16 + lh * 8]);
    }
    // B fragment (32x16 bf16): lane n, lanes 0-15 hold {k 0-7, k 16-23}, lanes 16-31 {k 8-15, k 24-31}.
    Frag fb[4];
#pragma unroll
    for (int j = 0; j < 4; ++j) {
      int n = wn + j * 16 + lrow;
      fb[j].u[0] = *reinterpret_cast<const v8us*>(&shB[buf][n * LDK + lh * 8]);
      fb[j].u[1] = *reinterpret_cast<const v8us*>(&shB[buf][n * LDK + 16 + lh * 8]);
    }
#pragma unroll
    for (int i = 0; i < 2; ++i)
#pragma unroll
      for (int j = 0; j < 4; ++j)
        acc[i][j] = __builtin_amdgcn_wmma_f32_16x16x32_bf16(
            false, fa[i].v, false, fb[j].v, (short)0, acc[i][j], false, false);
  };

  // Software pipeline: prologue issue, branch-free steady state, peeled tail.
  issue_tile(0);
  int cur = 0;
  for (int k0 = 0; k0 < K - TK; k0 += TK) {
    issue_tile(cur ^ 1);
    wait_asynccnt<4>();   // retire current tile's 4 copies (in-order)
    __syncthreads();      // all waves' copies for buffer `cur` visible
    compute_tile(cur);
    __syncthreads();      // done reading `cur` before it is refilled
    cur ^= 1;
  }
  wait_asynccnt<0>();
  __syncthreads();
  compute_tile(cur);

  // Epilogue. C fragment layout: VGPR r -> row (lh*8 + r), lane&15 -> col.
#pragma unroll
  for (int i = 0; i < 2; ++i) {
#pragma unroll
    for (int j = 0; j < 4; ++j) {
      const int mb = tileM + wm + i * 16 + lh * 8;
      const int nb = tileN + wn + j * 16 + lrow;
      if constexpr (EPI == EPI_COS) {
        float nn = nrm[nb];
#pragma unroll
        for (int r = 0; r < 8; ++r) {
          int m = mb + r;
          float v = acc[i][j][r] / (nrm[m] * nn + 1e-6f);
          Cf[(size_t)m * ldc + nb] = v;
        }
      } else if constexpr (EPI == EPI_TBF) {
        v8us pk;
#pragma unroll
        for (int r = 0; r < 8; ++r) pk[r] = f2bf_bits(acc[i][j][r]);
        *reinterpret_cast<v8us*>(Cb + (size_t)nb * ldcb + mb) = pk;
      } else if constexpr (EPI == EPI_BRELU) {
        float bv = bias[nb];
#pragma unroll
        for (int r = 0; r < 8; ++r) {
          int m = mb + r;
          float v = acc[i][j][r] + bv;
          v = v > 0.f ? v : 0.f;
          Cb[(size_t)m * ldcb + nb] = (bf16)v;
        }
      } else if constexpr (EPI == EPI_BRES) {
        float bv = bias[nb];
#pragma unroll
        for (int r = 0; r < 8; ++r) {
          int m = mb + r;
          float v = acc[i][j][r] + bv + res[(size_t)m * ldres + nb];
          Cb[(size_t)m * ldcb + nb] = (bf16)v;
        }
      } else {  // EPI_HEAD
        if (nb < nlim) {
          float bv = bias[nb];
#pragma unroll
          for (int r = 0; r < 8; ++r) {
            int m = mb + r;
            Cf[(size_t)m * ldc + nb] = acc[i][j][r] + bv;
          }
        }
      }
    }
  }
}

// ---------------------------------------------------------------------------
// Support kernels
// ---------------------------------------------------------------------------

// bf16 copy of features + row L2 norms (one block per row)
__global__ __launch_bounds__(256) void feat_prep_kernel(
    const float* __restrict__ x, bf16* __restrict__ xb, float* __restrict__ ln, int D) {
  int row = blockIdx.x;
  const float* xr = x + (size_t)row * D;
  bf16* xbr = xb + (size_t)row * D;
  float s = 0.f;
  for (int d = threadIdx.x; d < D; d += blockDim.x) {
    float v = xr[d];
    xbr[d] = (bf16)v;
    s += v * v;
  }
  __shared__ float red[256];
  red[threadIdx.x] = s;
  __syncthreads();
  for (int off = 128; off; off >>= 1) {
    if (threadIdx.x < off) red[threadIdx.x] += red[threadIdx.x + off];
    __syncthreads();
  }
  if (threadIdx.x == 0) ln[row] = sqrtf(red[0]);
}

// wt[n][k] = bf16(w[k][n]), n padded to Npad with zeros
__global__ void wt_bf_kernel(const float* __restrict__ w, bf16* __restrict__ wt,
                             int K, int N, int Npad) {
  long long i = (long long)blockIdx.x * blockDim.x + threadIdx.x;
  if (i >= (long long)Npad * K) return;
  int n = (int)(i / K), k = (int)(i % K);
  float v = (n < N) ? w[(size_t)k * N + n] : 0.f;
  wt[i] = (bf16)v;
}

// combined comp/reg head weights, transposed + padded to 128 rows x K
__global__ void wt_head_comp_kernel(const float* __restrict__ wc, const float* __restrict__ wr,
                                    bf16* __restrict__ wt, int K) {
  long long i = (long long)blockIdx.x * blockDim.x + threadIdx.x;
  if (i >= (long long)128 * K) return;
  int n = (int)(i / K), k = (int)(i % K);
  float v = 0.f;
  if (n < 20)      v = wc[(size_t)k * 20 + n];
  else if (n < 60) v = wr[(size_t)k * 40 + (n - 20)];
  wt[i] = (bf16)v;
}

__global__ void pack_cbias_kernel(const float* __restrict__ cb, const float* __restrict__ rb,
                                  float* __restrict__ dst) {
  int t = threadIdx.x;
  if (t < 20) dst[t] = cb[t];
  else if (t < 60) dst[t] = rb[t - 20];
  else if (t < 64) dst[t] = 0.f;
}

// Graph construction, one wave32 per (b,p) row. Top-6 nearest dis, top-2 sem,
// iou threshold, indicator normalization: mask = ind/(cnt+1e-6) + eye.
// Arg-min/max reductions tie-break on smaller index (matches jax top_k order).
__global__ __launch_bounds__(256) void build_mask_kernel(
    const float* __restrict__ iou, const float* __restrict__ dis,
    const float* __restrict__ coss, const int* __restrict__ props,
    float* __restrict__ mask, int P) {
  const int lane = threadIdx.x & 31;
  const int row = blockIdx.x * 8 + (threadIdx.x >> 5);
  const int b = row / P, p = row % P;
  const int pn = props[b];
  const bool pv = p < pn;
  const size_t base = (size_t)row * P;
  const float* ir = iou + base;
  const float* dr = dis + base;
  const float* cr = coss + base;
  float* mr = mask + base;

  int di[6];
#pragma unroll
  for (int t = 0; t < 6; ++t) {
    float best = 3.0e38f;
    int bi = P;
    for (int q = lane; q < P; q += 32) {
      bool skip = false;
      for (int u = 0; u < t; ++u) skip |= (di[u] == q);
      if (skip) continue;
      float v = (pv && q < pn) ? ((ir[q] > 0.f) ? 2.0f : dr[q]) : 1e9f;
      if (v < best) { best = v; bi = q; }
    }
#pragma unroll
    for (int o = 16; o; o >>= 1) {
      float ov = __shfl_xor(best, o, 32);
      int   oi = __shfl_xor(bi, o, 32);
      if (ov < best || (ov == best && oi < bi)) { best = ov; bi = oi; }
    }
    di[t] = bi;
  }

  int si[2];
#pragma unroll
  for (int t = 0; t < 2; ++t) {
    float best = -3.0e38f;
    int bi = P;
    for (int q = lane; q < P; q += 32) {
      if (t == 1 && si[0] == q) continue;
      float v = (pv && q < pn) ? ((ir[q] > 0.f) ? 0.f : (cr[q] - (q == p ? 1.f : 0.f)))
                               : -1e9f;
      if (v > best) { best = v; bi = q; }
    }
#pragma unroll
    for (int o = 16; o; o >>= 1) {
      float ov = __shfl_xor(best, o, 32);
      int   oi = __shfl_xor(bi, o, 32);
      if (ov > best || (ov == best && oi < bi)) { best = ov; bi = oi; }
    }
    si[t] = bi;
  }

  float cnt = 0.f;
  for (int q = lane; q < P; q += 32) {
    float iv = ir[q] - (q == p ? 1.f : 0.f);
    float temp = (iv > 0.7f) ? iv : 0.f;
    bool vv = pv && (q < pn);
    bool ni = !(ir[q] > 0.f);
    bool memd = (q == di[0]) | (q == di[1]) | (q == di[2]) | (q == di[3]) | (q == di[4]) |
                (q == di[5]);
    if (memd && ni && vv) temp += 1.f;
    bool mems = (q == si[0]) | (q == si[1]);
    if (mems && ni && vv) temp += 1.f;
    float ind = temp > 0.f ? 1.f : 0.f;
    cnt += ind;
    mr[q] = ind;
  }
#pragma unroll
  for (int o = 16; o; o >>= 1) cnt += __shfl_xor(cnt, o, 32);
  float inv = 1.f / (cnt + 1e-6f);
  for (int q = lane; q < P; q += 32) mr[q] = mr[q] * inv + (q == p ? 1.f : 0.f);
}

// adj = bf16(relu(cos * mask))
__global__ void make_adj_kernel(const float* __restrict__ coss, const float* __restrict__ mask,
                                bf16* __restrict__ adj, long long n) {
  long long i = (long long)blockIdx.x * blockDim.x + threadIdx.x;
  if (i >= n) return;
  float v = coss[i] * mask[i];
  adj[i] = (bf16)(v > 0.f ? v : 0.f);
}

// ---------------------------------------------------------------------------
// Host launcher
// ---------------------------------------------------------------------------

extern "C" void kernel_launch(void* const* d_in, const int* in_sizes, int n_in,
                              void* d_out, int out_size, void* d_ws, size_t ws_size,
                              hipStream_t stream) {
  constexpr int B = 4, P = 1024, DA = 1024, DC = 3072, HID = 512;
  typedef long long ll;

  const float* act_fts  = (const float*)d_in[0];
  const float* comp_fts = (const float*)d_in[1];
  const float* iou      = (const float*)d_in[2];
  const float* dis      = (const float*)d_in[3];
  const int*   props    = (const int*)d_in[4];
  const float* a_w1 = (const float*)d_in[5];
  const float* a_b1 = (const float*)d_in[6];
  const float* a_w2 = (const float*)d_in[7];
  const float* a_b2 = (const float*)d_in[8];
  const float* c_w1 = (const float*)d_in[9];
  const float* c_b1 = (const float*)d_in[10];
  const float* c_w2 = (const float*)d_in[11];
  const float* c_b2 = (const float*)d_in[12];
  const float* fc_act_w  = (const float*)d_in[13];
  const float* fc_act_b  = (const float*)d_in[14];
  const float* fc_comp_w = (const float*)d_in[15];
  const float* fc_comp_b = (const float*)d_in[16];
  const float* fc_reg_w  = (const float*)d_in[17];
  const float* fc_reg_b  = (const float*)d_in[18];
  float* out = (float*)d_out;

  char* wsb = (char*)d_ws;
  size_t off = 0;
  auto alloc = [&](size_t bytes) -> void* {
    void* p = wsb + off;
    off = (off + bytes + 255) & ~(size_t)255;
    return p;
  };

  bf16* act_bf  = (bf16*)alloc((size_t)B * P * DA * 2);
  bf16* comp_bf = (bf16*)alloc((size_t)B * P * DC * 2);
  float* act_ln  = (float*)alloc((size_t)B * P * 4);
  float* comp_ln = (float*)alloc((size_t)B * P * 4);
  float* act_cos  = (float*)alloc((size_t)B * P * P * 4);
  float* comp_cos = (float*)alloc((size_t)B * P * P * 4);
  float* maskb    = (float*)alloc((size_t)B * P * P * 4);
  bf16* act_adj  = (bf16*)alloc((size_t)B * P * P * 2);
  bf16* comp_adj = (bf16*)alloc((size_t)B * P * P * 2);
  bf16* a_w1t = (bf16*)alloc((size_t)HID * DA * 2);
  bf16* a_w2t = (bf16*)alloc((size_t)DA * HID * 2);
  bf16* c_w1t = (bf16*)alloc((size_t)HID * DC * 2);
  bf16* c_w2t = (bf16*)alloc((size_t)DC * HID * 2);
  bf16* fca_wt = (bf16*)alloc((size_t)128 * DA * 2);   // padded to 128 rows
  bf16* fcc_wt = (bf16*)alloc((size_t)128 * DC * 2);   // padded to 128 rows
  float* cbias = (float*)alloc(64 * 4);
  bf16* t0aT = (bf16*)alloc((size_t)B * HID * P * 2);
  bf16* h1a  = (bf16*)alloc((size_t)B * P * HID * 2);
  bf16* t1aT = (bf16*)alloc((size_t)B * DA * P * 2);
  bf16* act_out_bf = (bf16*)alloc((size_t)B * P * DA * 2);
  bf16* t0cT = (bf16*)alloc((size_t)B * HID * P * 2);
  bf16* h1c  = (bf16*)alloc((size_t)B * P * HID * 2);
  bf16* t1cT = (bf16*)alloc((size_t)B * DC * P * 2);
  bf16* comp_out_bf = (bf16*)alloc((size_t)B * P * DC * 2);
  (void)ws_size; (void)in_sizes; (void)n_in; (void)out_size;

  dim3 blk(256);
  auto gg = [](int M, int N) { return dim3((N + TN - 1) / TN, (M + TM - 1) / TM, B); };

  // --- prep ---
  feat_prep_kernel<<<B * P, 256, 0, stream>>>(act_fts, act_bf, act_ln, DA);
  feat_prep_kernel<<<B * P, 256, 0, stream>>>(comp_fts, comp_bf, comp_ln, DC);
  wt_bf_kernel<<<(HID * DA + 255) / 256, 256, 0, stream>>>(a_w1, a_w1t, DA, HID, HID);
  wt_bf_kernel<<<(DA * HID + 255) / 256, 256, 0, stream>>>(a_w2, a_w2t, HID, DA, DA);
  wt_bf_kernel<<<(HID * DC + 255) / 256, 256, 0, stream>>>(c_w1, c_w1t, DC, HID, HID);
  wt_bf_kernel<<<(DC * HID + 255) / 256, 256, 0, stream>>>(c_w2, c_w2t, HID, DC, DC);
  wt_bf_kernel<<<(128 * DA + 255) / 256, 256, 0, stream>>>(fc_act_w, fca_wt, DA, 21, 128);
  wt_head_comp_kernel<<<(128 * DC + 255) / 256, 256, 0, stream>>>(fc_comp_w, fc_reg_w, fcc_wt, DC);
  pack_cbias_kernel<<<1, 64, 0, stream>>>(fc_comp_b, fc_reg_b, cbias);

  // --- cosine-similarity grams (WMMA) ---
  gemm_bf16_wmma<EPI_COS><<<gg(P, P), blk, 0, stream>>>(
      act_bf, (ll)P * DA, DA, act_bf, (ll)P * DA, DA, P, P, DA,
      act_cos, (ll)P * P, P, nullptr, 0, 0, nullptr, nullptr, 0, 0, act_ln, P, 0);
  gemm_bf16_wmma<EPI_COS><<<gg(P, P), blk, 0, stream>>>(
      comp_bf, (ll)P * DC, DC, comp_bf, (ll)P * DC, DC, P, P, DC,
      comp_cos, (ll)P * P, P, nullptr, 0, 0, nullptr, nullptr, 0, 0, comp_ln, P, 0);

  // --- graph construction ---
  build_mask_kernel<<<(B * P) / 8, 256, 0, stream>>>(iou, dis, act_cos, props, maskb, P);
  ll npp = (ll)B * P * P;
  make_adj_kernel<<<(unsigned)((npp + 255) / 256), 256, 0, stream>>>(act_cos, maskb, act_adj, npp);
  make_adj_kernel<<<(unsigned)((npp + 255) / 256), 256, 0, stream>>>(comp_cos, maskb, comp_adj, npp);

  // --- Act GCN ---
  gemm_bf16_wmma<EPI_TBF><<<gg(P, HID), blk, 0, stream>>>(
      act_bf, (ll)P * DA, DA, a_w1t, 0, DA, P, HID, DA,
      nullptr, 0, 0, t0aT, (ll)HID * P, P, nullptr, nullptr, 0, 0, nullptr, 0, 0);
  gemm_bf16_wmma<EPI_BRELU><<<gg(P, HID), blk, 0, stream>>>(
      act_adj, (ll)P * P, P, t0aT, (ll)HID * P, P, P, HID, P,
      nullptr, 0, 0, h1a, (ll)P * HID, HID, a_b1, nullptr, 0, 0, nullptr, 0, 0);
  gemm_bf16_wmma<EPI_TBF><<<gg(P, DA), blk, 0, stream>>>(
      h1a, (ll)P * HID, HID, a_w2t, 0, HID, P, DA, HID,
      nullptr, 0, 0, t1aT, (ll)DA * P, P, nullptr, nullptr, 0, 0, nullptr, 0, 0);
  gemm_bf16_wmma<EPI_BRES><<<gg(P, DA), blk, 0, stream>>>(
      act_adj, (ll)P * P, P, t1aT, (ll)DA * P, P, P, DA, P,
      nullptr, 0, 0, act_out_bf, (ll)P * DA, DA, a_b2, act_fts, (ll)P * DA, DA, nullptr, 0, 0);

  // --- Comp GCN ---
  gemm_bf16_wmma<EPI_TBF><<<gg(P, HID), blk, 0, stream>>>(
      comp_bf, (ll)P * DC, DC, c_w1t, 0, DC, P, HID, DC,
      nullptr, 0, 0, t0cT, (ll)HID * P, P, nullptr, nullptr, 0, 0, nullptr, 0, 0);
  gemm_bf16_wmma<EPI_BRELU><<<gg(P, HID), blk, 0, stream>>>(
      comp_adj, (ll)P * P, P, t0cT, (ll)HID * P, P, P, HID, P,
      nullptr, 0, 0, h1c, (ll)P * HID, HID, c_b1, nullptr, 0, 0, nullptr, 0, 0);
  gemm_bf16_wmma<EPI_TBF><<<gg(P, DC), blk, 0, stream>>>(
      h1c, (ll)P * HID, HID, c_w2t, 0, HID, P, DC, HID,
      nullptr, 0, 0, t1cT, (ll)DC * P, P, nullptr, nullptr, 0, 0, nullptr, 0, 0);
  gemm_bf16_wmma<EPI_BRES><<<gg(P, DC), blk, 0, stream>>>(
      comp_adj, (ll)P * P, P, t1cT, (ll)DC * P, P, P, DC, P,
      nullptr, 0, 0, comp_out_bf, (ll)P * DC, DC, c_b2, comp_fts, (ll)P * DC, DC, nullptr, 0, 0);

  // --- heads: out is (B*P, 81) = [act(21) | comp(20) | reg(40)] ---
  gemm_bf16_wmma<EPI_HEAD><<<gg(P, 128), blk, 0, stream>>>(
      act_out_bf, (ll)P * DA, DA, fca_wt, 0, DA, P, 128, DA,
      out, (ll)P * 81, 81, nullptr, 0, 0, fc_act_b, nullptr, 0, 0, nullptr, 0, 21);
  gemm_bf16_wmma<EPI_HEAD><<<gg(P, 128), blk, 0, stream>>>(
      comp_out_bf, (ll)P * DC, DC, fcc_wt, 0, DC, P, 128, DC,
      out + 21, (ll)P * 81, 81, nullptr, 0, 0, cbias, nullptr, 0, 0, nullptr, 0, 60);
}
